// F3Net_3599182594812
// MI455X (gfx1250) — matrix operand order repaired and checked by reading, not compiled, not run on hardware
//
#include <hip/hip_runtime.h>
#include <hip/hip_bf16.h>
#include <math.h>

// ---------------------------------------------------------------------------
// CDNA5 (gfx1250) F3Net forward.  Core engine: implicit-GEMM conv using
// v_wmma_f32_16x16x32_f16 (wave32).  GEMM K-order is channel-minor
// (k = rs*Cin + ci) so for power-of-two Cin the im2col decode is shift/mask
// only; (r,s) carried incrementally so the steady-state K-loop has no integer
// division at all.  f32 -> f16 at LDS staging (vectorized ds_store_b128),
// f32 accumulation, fused BN/bias + ReLU/sigmoid epilogue.
// Workgroup: 256 thr = 8 waves; tile M=64 x N=256; wave tile 32x64 = 8 WMMA.
// ---------------------------------------------------------------------------

typedef __attribute__((ext_vector_type(16))) _Float16 v16h;
typedef __attribute__((ext_vector_type(8)))  _Float16 v8h;
typedef __attribute__((ext_vector_type(8)))  float    v8f;

#define KSTEP 32
#define LDSTR 40   // 32 halves + 8 pad; rows = 80B so 16B-aligned frag offsets

// Assemble v16h fragment from two 16-byte LDS reads.
__device__ inline v16h frag16(const _Float16* p0, const _Float16* p1)
{
    v8h t0 = *(const v8h*)p0;
    v8h t1 = *(const v8h*)p1;
    v16h r;
#pragma unroll
    for (int i = 0; i < 8; ++i) { r[i] = t0[i]; r[i + 8] = t1[i]; }
    return r;
}

// C/D tile store: 16x16 f32, VGPR r holds M = r + 8*(lane>=16), N = lane%16.
__device__ inline void store_tile(const v8f& c, int mbase, int nbase,
                                  float* __restrict__ y,
                                  const float* __restrict__ scale,
                                  const float* __restrict__ shift,
                                  int act, int Cout, int HoWo, int Npix)
{
    const int lane = threadIdx.x & 31;
    const int hh   = lane >> 4;
    const int lr   = lane & 15;
    const int n    = nbase + lr;
    if (n >= Npix) return;
    const int b   = n / HoWo;
    const int rem = n - b * HoWo;
    float* yo = y + (size_t)b * Cout * HoWo + rem;
#pragma unroll
    for (int r = 0; r < 8; ++r) {
        const int m = mbase + r + hh * 8;
        if (m >= Cout) continue;
        float v = c[r];
        if (scale) v *= scale[m];
        if (shift) v += shift[m];
        if (act == 1)      v = v > 0.f ? v : 0.f;
        else if (act == 2) v = 1.f / (1.f + expf(-v));
        yo[(size_t)m * HoWo] = v;
    }
}

__global__ __launch_bounds__(256)
void conv_wmma(const float* __restrict__ x, const float* __restrict__ w,
               const float* __restrict__ scale, const float* __restrict__ shift,
               float* __restrict__ y,
               int Bn, int Cin, int H, int W, int Cout, int R, int Sk,
               int stride, int pad, int Ho, int Wo, int act, int cshift)
{
    __shared__ __attribute__((aligned(16))) _Float16 Alds[64  * LDSTR];
    __shared__ __attribute__((aligned(16))) _Float16 Blds[256 * LDSTR];

    const int tid  = threadIdx.x;
    const int wave = tid >> 5;
    const int lane = tid & 31;
    const int hh   = lane >> 4;
    const int lr   = lane & 15;

    const int HoWo = Ho * Wo;
    const int Npix = Bn * HoWo;       // <= 1.6M, int32 is fine
    const int RS   = R * Sk;
    const int Ktot = Cin * RS;
    const int HW   = H * W;

    const int m0 = blockIdx.y * 64;
    const int n0 = blockIdx.x * 256;

    // ---- B staging: thread tid stages the full 32-half row for pixel n0+tid
    const int  npix   = n0 + tid;
    const bool pvalid = npix < Npix;
    int bb = 0, oh = 0, ow = 0;
    if (pvalid) {
        bb = npix / HoWo;
        const int rem = npix - bb * HoWo;
        oh = rem / Wo; ow = rem - oh * Wo;
    }
    const int ih0 = oh * stride - pad;
    const int iw0 = ow * stride - pad;
    const float* xin = x + (size_t)bb * Cin * HW;
    _Float16* bdst = &Blds[tid * LDSTR];

    // ---- A staging: thread -> weight row rA (tid>>2), 8 cols at (tid&3)*8
    const int rA = tid >> 2;
    const int cA = (tid & 3) * 8;
    const int mA = m0 + rA;
    _Float16* adst = &Alds[rA * LDSTR + cA];
    const bool mvalid = mA < Cout;

    // wave sub-tile (32 x 64) within the 64 x 256 workgroup tile
    const int wm = (wave & 1) * 32;
    const int wn = (wave >> 1) * 64;

    v8f acc[2][4] = {};

    // incremental (r,s) decomposition of rs = kk >> cshift (fast path only)
    int rr = 0, ss = 0;

    for (int kk = 0; kk < Ktot; kk += KSTEP) {
        if (cshift >= 0) {
            // ======== fast path: Cin = pow2 >= 32 (no partial K-blocks) ====
            const int rs  = kk >> cshift;            // uniform over block
            const int ci0 = kk & (Cin - 1);
            // A: w[mA*Ktot + (ci0+cA+i)*RS + rs], stride RS
            {
                const float* wr = w + (size_t)mA * Ktot + (size_t)(ci0 + cA) * RS + rs;
                if (kk + KSTEP < Ktot) __builtin_prefetch(wr + 32 * RS, 0, 0);
                v8h t;
#pragma unroll
                for (int i = 0; i < 8; ++i)
                    t[i] = (_Float16)(mvalid ? wr[i * RS] : 0.f);
                *(v8h*)adst = t;                     // ds_store_b128
            }
            // B: x[bb, ci0+i, ih0+rr, iw0+ss], stride HW; one bounds check
            {
                const int ih = ih0 + rr, iw = iw0 + ss;
                if (pvalid && (unsigned)ih < (unsigned)H && (unsigned)iw < (unsigned)W) {
                    const float* src = xin + ((size_t)ci0 * H + ih) * W + iw;
#pragma unroll
                    for (int q = 0; q < 4; ++q) {
                        v8h t;
#pragma unroll
                        for (int i = 0; i < 8; ++i)
                            t[i] = (_Float16)src[(size_t)(q * 8 + i) * HW];
                        *(v8h*)(bdst + q * 8) = t;   // ds_store_b128
                    }
                } else {
                    const v8h z = {};
#pragma unroll
                    for (int q = 0; q < 4; ++q)
                        *(v8h*)(bdst + q * 8) = z;
                }
            }
            // advance rs decomposition when next block crosses a Cin boundary
            if (((kk + KSTEP) & (Cin - 1)) == 0) {
                if (++ss == Sk) { ss = 0; ++rr; }
            }
        } else {
            // ======== generic path (Cin==3: stem / freq0, tiny Ktot) =======
            {
                v8h t;
#pragma unroll
                for (int i = 0; i < 8; ++i) {
                    const int k = kk + cA + i;
                    float v = 0.f;
                    if (mvalid && k < Ktot) {
                        const int rs = k / Cin;
                        const int ci = k - rs * Cin;
                        v = w[(size_t)mA * Ktot + (size_t)ci * RS + rs];
                    }
                    t[i] = (_Float16)v;
                }
                *(v8h*)adst = t;
            }
#pragma unroll
            for (int q = 0; q < 4; ++q) {
                v8h t;
#pragma unroll
                for (int i = 0; i < 8; ++i) {
                    const int k = kk + q * 8 + i;
                    float v = 0.f;
                    if (pvalid && k < Ktot) {
                        const int rs = k / Cin;
                        const int ci = k - rs * Cin;
                        const int r  = rs / Sk;
                        const int s  = rs - r * Sk;
                        const int ih = ih0 + r, iw = iw0 + s;
                        if ((unsigned)ih < (unsigned)H && (unsigned)iw < (unsigned)W)
                            v = xin[((size_t)ci * H + ih) * W + iw];
                    }
                    t[i] = (_Float16)v;
                }
                *(v8h*)(bdst + q * 8) = t;
            }
        }
        __syncthreads();

        // ---- fragments (A and B both use contiguous k = 16*hh + s) ----
        v16h af[2], bf[4];
#pragma unroll
        for (int mi = 0; mi < 2; ++mi) {
            const _Float16* ap = &Alds[(wm + mi * 16 + lr) * LDSTR];
            af[mi] = frag16(ap + hh * 16, ap + hh * 16 + 8);
        }
#pragma unroll
        for (int nj = 0; nj < 4; ++nj) {
            const _Float16* bp = &Blds[(wn + nj * 16 + lr) * LDSTR + hh * 16];
            bf[nj] = frag16(bp, bp + 8);
        }

#pragma unroll
        for (int mi = 0; mi < 2; ++mi)
#pragma unroll
            for (int nj = 0; nj < 4; ++nj)
                acc[mi][nj] = __builtin_amdgcn_wmma_f32_16x16x32_f16(
                    false, af[mi], false, bf[nj], (short)0, acc[mi][nj], false, false);

        __syncthreads();
    }

#pragma unroll
    for (int mi = 0; mi < 2; ++mi)
#pragma unroll
        for (int nj = 0; nj < 4; ++nj)
            store_tile(acc[mi][nj], m0 + wm + mi * 16, n0 + wn + nj * 16,
                       y, scale, shift, act, Cout, HoWo, Npix);
}

// ---------------------------------------------------------------------------
// Blockwise 8x8 DCT-II (Y = D X D^T) + high-pass mask. 4 blocks per workgroup.
// ---------------------------------------------------------------------------
__global__ void dct_mask_kernel(const float* __restrict__ x, const float* __restrict__ D,
                                float* __restrict__ out, int BC, int Hb, int Wb)
{
    __shared__ float Ds[64];
    __shared__ float Xs[4][64];
    __shared__ float Ts[4][64];
    const int tid = threadIdx.x;
    if (tid < 64) Ds[tid] = D[tid];
    const int sub = tid >> 6;
    const int t   = tid & 63;
    const int i   = t >> 3, j = t & 7;
    const int gblk = blockIdx.x * 4 + sub;
    const int nblk = BC * Hb * Wb;
    const bool valid = gblk < nblk;
    int bw = 0, bh = 0, bc = 0;
    if (valid) {
        bw = gblk % Wb;
        const int r = gblk / Wb;
        bh = r % Hb;
        bc = r / Hb;
    }
    const int H = Hb * 8, W = Wb * 8;
    const float* xp = x + ((size_t)bc * H + (size_t)bh * 8) * W + (size_t)bw * 8;
    Xs[sub][t] = valid ? xp[i * W + j] : 0.f;
    __syncthreads();
    float acc = 0.f;
#pragma unroll
    for (int l = 0; l < 8; ++l) acc += Ds[i * 8 + l] * Xs[sub][l * 8 + j];
    Ts[sub][t] = acc;
    __syncthreads();
    float acc2 = 0.f;
#pragma unroll
    for (int l = 0; l < 8; ++l) acc2 += Ts[sub][i * 8 + l] * Ds[j * 8 + l];
    const int h = bh * 8 + i, w2 = bw * 8 + j;
    const float mask = (h < H / 4 && w2 < W / 4) ? 0.1f : 1.0f;
    if (valid) out[((size_t)bc * H + h) * W + w2] = acc2 * mask;
}

// ---------------------------------------------------------------------------
__global__ void bn_prep(const float* __restrict__ g, const float* __restrict__ b,
                        const float* __restrict__ m, const float* __restrict__ v,
                        float* __restrict__ scale, float* __restrict__ shift, int C)
{
    const int i = blockIdx.x * blockDim.x + threadIdx.x;
    if (i >= C) return;
    const float inv = rsqrtf(v[i] + 1e-5f);
    const float sc  = g[i] * inv;
    scale[i] = sc;
    shift[i] = b[i] - m[i] * sc;
}

__global__ void maxpool3s2(const float* __restrict__ in, float* __restrict__ out,
                           int BC, int H, int W, int Ho, int Wo)
{
    const int idx = blockIdx.x * blockDim.x + threadIdx.x;
    const int tot = BC * Ho * Wo;
    if (idx >= tot) return;
    const int ow = idx % Wo;
    const int r  = idx / Wo;
    const int oh = r % 56;   // Ho == 56 here, but keep generic below
    const int oh2 = r % Ho;
    const int bc = r / Ho;
    (void)oh;
    const float* ip = in + (size_t)bc * H * W;
    float m = -1e30f;
    for (int dy = -1; dy <= 1; ++dy)
        for (int dx = -1; dx <= 1; ++dx) {
            const int ih = oh2 * 2 + dy, iw = ow * 2 + dx;
            if ((unsigned)ih < (unsigned)H && (unsigned)iw < (unsigned)W)
                m = fmaxf(m, ip[ih * W + iw]);
        }
    out[idx] = m;
}

__global__ void se_reduce(const float* __restrict__ f, float* __restrict__ avg,
                          float* __restrict__ mx, int BC, int HW)
{
    const int idx = blockIdx.x * blockDim.x + threadIdx.x;
    if (idx >= BC) return;
    const float* p = f + (size_t)idx * HW;
    float s = 0.f, m = -1e30f;
    for (int i = 0; i < HW; ++i) { const float v = p[i]; s += v; m = fmaxf(m, v); }
    avg[idx] = s / HW;
    mx[idx]  = m;
}

__global__ void se_hidden(const float* __restrict__ v, const float* __restrict__ W1,
                          float* __restrict__ h, int Bn, int C, int Hd)
{   // h[b][j] = relu(sum_c v[b][c] * W1[j][c])
    const int idx = blockIdx.x * blockDim.x + threadIdx.x;
    if (idx >= Bn * Hd) return;
    const int b = idx / Hd, j = idx % Hd;
    const float* vp = v + (size_t)b * C;
    const float* wp = W1 + (size_t)j * C;
    float s = 0.f;
    for (int c = 0; c < C; ++c) s += vp[c] * wp[c];
    h[idx] = s > 0.f ? s : 0.f;
}

__global__ void se_att(const float* __restrict__ ha, const float* __restrict__ hm,
                       const float* __restrict__ W2, float* __restrict__ att,
                       int Bn, int C, int Hd)
{   // att[b][c] = sigmoid(sum_j (ha+hm)[b][j] * W2[c][j])
    const int idx = blockIdx.x * blockDim.x + threadIdx.x;
    if (idx >= Bn * C) return;
    const int b = idx / C, c = idx % C;
    const float* wp = W2 + (size_t)c * Hd;
    float s = 0.f;
    for (int j = 0; j < Hd; ++j) s += (ha[b * Hd + j] + hm[b * Hd + j]) * wp[j];
    att[idx] = 1.f / (1.f + expf(-s));
}

__global__ void se_scale(float* __restrict__ f, const float* __restrict__ att,
                         int BC, int HW)
{
    const int idx = blockIdx.x * blockDim.x + threadIdx.x;
    const int tot = BC * HW;
    if (idx >= tot) return;
    f[idx] *= att[idx / HW];
}

__global__ void pool_concat(const float* __restrict__ s, const float* __restrict__ f,
                            float* __restrict__ h, int Bn, int Cs, int HWs, int Cf, int HWf)
{
    const int idx = blockIdx.x * blockDim.x + threadIdx.x;
    const int Ct = Cs + Cf;
    if (idx >= Bn * Ct) return;
    const int b = idx / Ct, c = idx % Ct;
    float sum = 0.f;
    if (c < Cs) {
        const float* p = s + ((size_t)b * Cs + c) * HWs;
        for (int i = 0; i < HWs; ++i) sum += p[i];
        sum /= HWs;
    } else {
        const float* p = f + ((size_t)b * Cf + (c - Cs)) * HWf;
        for (int i = 0; i < HWf; ++i) sum += p[i];
        sum /= HWf;
    }
    h[idx] = sum;
}

// ---------------------------------------------------------------------------
extern "C" void kernel_launch(void* const* d_in, const int* in_sizes, int n_in,
                              void* d_out, int out_size, void* d_ws, size_t ws_size,
                              hipStream_t stream)
{
    (void)out_size; (void)ws_size;
    struct Pr {
        const float *x, *dct, *stem_w;
        const float *stem_bn[4];               // g,b,m,v
        const float *bw1[8], *bw2[8];
        const float *bbn1[8][4], *bbn2[8][4];  // g,b,m,v
        const float *fw[4], *fbn[4][4];
        const float *aw1, *aw2;
        const float *clsW[3], *clsb[3];
    } P;
    int ii = 0;
    auto nx = [&]() -> const float* { return (const float*)d_in[ii++]; };
    const bool insertion = (n_in > 0 && in_sizes[0] == 32 * 3 * 224 * 224);
    if (insertion) {   // dict-insertion order
        P.x = nx(); P.dct = nx(); P.stem_w = nx();
        for (int k = 0; k < 4; ++k) P.stem_bn[k] = nx();                  // g,b,m,v
        for (int b = 0; b < 8; ++b) {
            P.bw1[b] = nx();
            for (int k = 0; k < 4; ++k) P.bbn1[b][k] = nx();
            P.bw2[b] = nx();
            for (int k = 0; k < 4; ++k) P.bbn2[b][k] = nx();
        }
        for (int f = 0; f < 4; ++f) { P.fw[f] = nx(); for (int k = 0; k < 4; ++k) P.fbn[f][k] = nx(); }
        P.aw1 = nx(); P.aw2 = nx();
        for (int c = 0; c < 3; ++c) { P.clsW[c] = nx(); P.clsb[c] = nx(); }
    } else {           // jax pytree sorted-key order (bn dicts flatten b,g,m,v)
        P.aw1 = nx(); P.aw2 = nx();
        for (int b = 0; b < 8; ++b) {
            const float *bb, *gg, *mm, *vv;
            bb = nx(); gg = nx(); mm = nx(); vv = nx();
            P.bbn1[b][0] = gg; P.bbn1[b][1] = bb; P.bbn1[b][2] = mm; P.bbn1[b][3] = vv;
            bb = nx(); gg = nx(); mm = nx(); vv = nx();
            P.bbn2[b][0] = gg; P.bbn2[b][1] = bb; P.bbn2[b][2] = mm; P.bbn2[b][3] = vv;
            P.bw1[b] = nx(); P.bw2[b] = nx();
        }
        for (int c = 0; c < 3; ++c) { P.clsW[c] = nx(); P.clsb[c] = nx(); }
        P.dct = nx();
        for (int f = 0; f < 4; ++f) {
            const float *bb = nx(), *gg = nx(), *mm = nx(), *vv = nx();
            P.fbn[f][0] = gg; P.fbn[f][1] = bb; P.fbn[f][2] = mm; P.fbn[f][3] = vv;
            P.fw[f] = nx();
        }
        { const float *bb = nx(), *gg = nx(), *mm = nx(), *vv = nx();
          P.stem_bn[0] = gg; P.stem_bn[1] = bb; P.stem_bn[2] = mm; P.stem_bn[3] = vv; }
        P.stem_w = nx();
        P.x = nx();
    }

    // ---- workspace bump allocator ----
    char* wp = (char*)d_ws;
    auto alloc = [&](size_t nelem) -> float* {
        float* p = (float*)wp;
        wp += ((nelem * sizeof(float) + 255) & ~(size_t)255);
        return p;
    };
    float* bufP  = alloc((size_t)32 * 64 * 224 * 224);   // 411 MB ping
    float* bufQ  = alloc((size_t)32 * 64 * 224 * 224);   // 411 MB pong
    float* F0    = alloc((size_t)32 * 3  * 224 * 224);   // DCT output
    float* S1    = alloc((size_t)32 * 64 * 56  * 56);    // spatial ping
    float* d_scale = alloc(512);
    float* d_shift = alloc(512);
    float* avg = alloc(32 * 512); float* mx = alloc(32 * 512);
    float* ha  = alloc(32 * 32);  float* hm = alloc(32 * 32);
    float* att = alloc(32 * 512);
    float* hbuf = alloc(32 * 1024);
    float* h1   = alloc(32 * 512);
    float* h2   = alloc(32 * 128);

    auto conv = [&](const float* xin, const float* w, const float* sc, const float* sh,
                    float* y, int Bn, int Cin, int H, int W, int Cout,
                    int R, int S, int str, int pad, int act) {
        const int Ho = (H + 2 * pad - R) / str + 1;
        const int Wo = (W + 2 * pad - S) / str + 1;
        const int Npix = Bn * Ho * Wo;
        int cshift = -1;
        if (Cin >= 32 && (Cin & (Cin - 1)) == 0) {
            cshift = 0;
            while ((1 << cshift) < Cin) ++cshift;
        }
        dim3 g((unsigned)((Npix + 255) / 256), (unsigned)((Cout + 63) / 64));
        conv_wmma<<<g, 256, 0, stream>>>(xin, w, sc, sh, y, Bn, Cin, H, W, Cout,
                                         R, S, str, pad, Ho, Wo, act, cshift);
    };
    auto bnp = [&](const float* const bn[4], int C) {
        bn_prep<<<(C + 255) / 256, 256, 0, stream>>>(bn[0], bn[1], bn[2], bn[3],
                                                     d_scale, d_shift, C);
    };

    // ================= frequency branch =================
    dct_mask_kernel<<<(96 * 28 * 28 + 3) / 4, 256, 0, stream>>>(P.x, P.dct, F0, 96, 28, 28);
    bnp(P.fbn[0], 64);
    conv(F0,   P.fw[0], d_scale, d_shift, bufP, 32,   3, 224, 224,  64, 3, 3, 1, 1, 1);
    bnp(P.fbn[1], 128);
    conv(bufP, P.fw[1], d_scale, d_shift, bufQ, 32,  64, 224, 224, 128, 3, 3, 2, 1, 1);
    bnp(P.fbn[2], 256);
    conv(bufQ, P.fw[2], d_scale, d_shift, bufP, 32, 128, 112, 112, 256, 3, 3, 2, 1, 1);
    bnp(P.fbn[3], 512);
    conv(bufP, P.fw[3], d_scale, d_shift, bufQ, 32, 256,  56,  56, 512, 3, 3, 2, 1, 1);
    // f = bufQ : (32,512,28,28)

    // squeeze-excite attention
    se_reduce<<<(32 * 512 + 255) / 256, 256, 0, stream>>>(bufQ, avg, mx, 32 * 512, 28 * 28);
    se_hidden<<<(32 * 32 + 255) / 256, 256, 0, stream>>>(avg, P.aw1, ha, 32, 512, 32);
    se_hidden<<<(32 * 32 + 255) / 256, 256, 0, stream>>>(mx,  P.aw1, hm, 32, 512, 32);
    se_att<<<(32 * 512 + 255) / 256, 256, 0, stream>>>(ha, hm, P.aw2, att, 32, 512, 32);
    se_scale<<<(32 * 512 * 784 + 255) / 256, 256, 0, stream>>>(bufQ, att, 32 * 512, 784);

    // ================= spatial branch =================
    bnp(P.stem_bn, 64);
    conv(P.x, P.stem_w, d_scale, d_shift, bufP, 32, 3, 224, 224, 64, 7, 7, 2, 3, 1);
    maxpool3s2<<<(32 * 64 * 56 * 56 + 255) / 256, 256, 0, stream>>>(bufP, S1, 32 * 64, 112, 112, 56, 56);

    const int bin[8]  = { 64,  64,  64, 128, 128, 256, 256, 512};
    const int bout[8] = { 64,  64, 128, 128, 256, 256, 512, 512};
    const int bstr[8] = {  1,   1,   2,   1,   2,   1,   2,   1};
    int Hcur = 56;
    for (int b = 0; b < 8; ++b) {
        const int Hout = (Hcur + 2 - 3) / bstr[b] + 1;
        bnp(P.bbn1[b], bout[b]);
        conv(S1,   P.bw1[b], d_scale, d_shift, bufP, 32, bin[b],  Hcur, Hcur, bout[b], 3, 3, bstr[b], 1, 1);
        bnp(P.bbn2[b], bout[b]);
        conv(bufP, P.bw2[b], d_scale, d_shift, S1,   32, bout[b], Hout, Hout, bout[b], 3, 3, 1,       1, 0);
        Hcur = Hout;
    }
    // s = S1 : (32,512,7,7)

    // ================= head =================
    pool_concat<<<(32 * 1024 + 255) / 256, 256, 0, stream>>>(S1, bufQ, hbuf, 32, 512, 49, 512, 784);
    // classifier layers as 1x1 WMMA "convs" (scale=null -> identity, shift=bias)
    conv(hbuf, P.clsW[0], nullptr, P.clsb[0], h1,            32, 1024, 1, 1, 512, 1, 1, 1, 0, 1);
    conv(h1,   P.clsW[1], nullptr, P.clsb[1], h2,            32,  512, 1, 1, 128, 1, 1, 1, 0, 1);
    conv(h2,   P.clsW[2], nullptr, P.clsb[2], (float*)d_out, 32,  128, 1, 1,   1, 1, 1, 1, 0, 2);
}
